// DyGMamba_28527172780214
// MI455X (gfx1250) — matrix operand order repaired and checked by reading, not compiled, not run on hardware
//
#include <hip/hip_runtime.h>

typedef __attribute__((ext_vector_type(16))) __bf16 v16bf;
typedef __attribute__((ext_vector_type(8)))  float  v8f;
typedef __attribute__((ext_vector_type(8)))  unsigned short u16x8;
typedef __attribute__((ext_vector_type(4)))  float  f32x4;

static constexpr int BB = 256;
static constexpr int LL = 512;
static constexpr int FF = 64;
static constexpr int NTILES = BB * LL / 16;   // 8192 row-tiles of 16

struct U16A { unsigned short x[16]; };
struct U8x2 { u16x8 lo, hi; };

__device__ __forceinline__ unsigned short f2bf(float f) {
  union { float f; unsigned u; } v; v.f = f;
  unsigned r = v.u + 0x7fffu + ((v.u >> 16) & 1u);   // round-to-nearest-even
  return (unsigned short)(r >> 16);
}
__device__ __forceinline__ float bf2f(unsigned short h) {
  union { unsigned u; float f; } v; v.u = ((unsigned)h) << 16;
  return v.f;
}
__device__ __forceinline__ v16bf pack16(const unsigned short* s) {
  U16A u;
#pragma unroll
  for (int i = 0; i < 16; ++i) u.x[i] = s[i];
  return __builtin_bit_cast(v16bf, u);
}
// A-fragment: p -> [row][ks*32 + hi*8]; octets at +0 and +16 elements
__device__ __forceinline__ v16bf load_a16frag(const unsigned short* p) {
  U8x2 u;
  u.lo = *(const u16x8*)p;
  u.hi = *(const u16x8*)(p + 16);
  return __builtin_bit_cast(v16bf, u);
}
// B-fragment: p -> [col][ks*32 + hi*8]; octets at +0 and +16 elements
__device__ __forceinline__ v16bf load_b16frag(const unsigned short* p) {
  U8x2 u;
  u.lo = *(const u16x8*)p;
  u.hi = *(const u16x8*)(p + 16);
  return __builtin_bit_cast(v16bf, u);
}
__device__ __forceinline__ v8f wmma_bf16(v16bf a, v16bf b, v8f c) {
  return __builtin_amdgcn_wmma_f32_16x16x32_bf16(false, a, false, b, (short)0, c, false, false);
}

// ------------------------------------------------ weights -> bf16, TRANSPOSED [n][k]
__global__ __launch_bounds__(256)
void k_cvt_weights(const float* __restrict__ w2, const float* __restrict__ wq,
                   const float* __restrict__ wk, const float* __restrict__ wv,
                   const float* __restrict__ wo, unsigned short* __restrict__ outw) {
  int i = blockIdx.x * 256 + threadIdx.x;
  if (i < 4096) {
    int r = i >> 6, c = i & 63;        // W[r][c] -> WT[c][r]
    int o = c * 64 + r;
    outw[o]          = f2bf(w2[i]);
    outw[4096 + o]   = f2bf(wq[i]);
    outw[8192 + o]   = f2bf(wk[i]);
    outw[12288 + o]  = f2bf(wv[i]);
    outw[16384 + o]  = f2bf(wo[i]);
  }
}

// ---------------------------------------------------- histogram counts + encoder L1
__global__ __launch_bounds__(256)
void k_count_encode(const int* __restrict__ src, const int* __restrict__ dst,
                    const float* __restrict__ w1, const float* __restrict__ b1,
                    unsigned short* __restrict__ gsrc, unsigned short* __restrict__ gdst) {
  __shared__ int hs[2000];
  __shared__ int hd[2000];
  __shared__ float w1s[64], b1s[64];
  int b = blockIdx.x, tid = threadIdx.x;
  for (int i = tid; i < 2000; i += 256) { hs[i] = 0; hd[i] = 0; }
  if (tid < 64) { w1s[tid] = w1[tid]; b1s[tid] = b1[tid]; }
  __syncthreads();
  const int* sr = src + (size_t)b * LL;
  const int* dr = dst + (size_t)b * LL;
  for (int i = tid; i < LL; i += 256) { atomicAdd(&hs[sr[i]], 1); atomicAdd(&hd[dr[i]], 1); }
  __syncthreads();
  for (int i = tid; i < LL; i += 256) {
    int sid = sr[i], did = dr[i];
    float s0 = sid ? (float)hs[sid] : 0.f;   // counts_in(src, src)
    float s1 = sid ? (float)hd[sid] : 0.f;   // counts_in(src, dst)
    float d0 = did ? (float)hd[did] : 0.f;   // counts_in(dst, dst)
    float d1 = did ? (float)hs[did] : 0.f;   // counts_in(dst, src)
    size_t base = ((size_t)b * LL + i) * FF;
#pragma unroll
    for (int fg = 0; fg < 8; ++fg) {         // vectorized 16B stores
      u16x8 vs, vd;
#pragma unroll
      for (int j = 0; j < 8; ++j) {
        int f = fg * 8 + j;
        float w = w1s[f], bb = b1s[f];
        vs[j] = f2bf(fmaxf(fmaf(s0, w, bb), 0.f) + fmaxf(fmaf(s1, w, bb), 0.f));
        vd[j] = f2bf(fmaxf(fmaf(d0, w, bb), 0.f) + fmaxf(fmaf(d1, w, bb), 0.f));
      }
      *(u16x8*)(gsrc + base + fg * 8) = vs;
      *(u16x8*)(gdst + base + fg * 8) = vd;
    }
  }
}

// ------------------- generic Y[M,64] = X[M,64] @ W[64,64], W given TRANSPOSED [n][k]
template <bool XF32>
__global__ __launch_bounds__(256)
void k_gemm64(const void* __restrict__ Xv, const unsigned short* __restrict__ WT,
              const float* __restrict__ bias, float bscale,
              float* __restrict__ Yf, unsigned short* __restrict__ Ybf) {
  int tid = threadIdx.x, lane = tid & 31, wid = tid >> 5;
  int col = lane & 15, hi = lane >> 4;
  // B fragments: contiguous rows of WT, held in registers for the whole kernel
  v16bf Bf[2][4];
#pragma unroll
  for (int ks = 0; ks < 2; ++ks)
#pragma unroll
    for (int nt = 0; nt < 4; ++nt)
      Bf[ks][nt] = load_b16frag(WT + (size_t)(nt * 16 + col) * 64 + ks * 32 + hi * 8);
  float bc[4];
#pragma unroll
  for (int nt = 0; nt < 4; ++nt) bc[nt] = bias ? bias[nt * 16 + col] * bscale : 0.f;

  int gw = blockIdx.x * 8 + wid;
  int stride = gridDim.x * 8;
  for (int t = gw; t < NTILES; t += stride) {
    v16bf Af[2];
    if constexpr (XF32) {
      const float* X = (const float*)Xv + ((size_t)t * 16 + col) * 64;
      unsigned short tmp[16];
#pragma unroll
      for (int ks = 0; ks < 2; ++ks) {
#pragma unroll
        for (int i = 0; i < 8; ++i) {
          tmp[i]     = f2bf(X[ks * 32 + hi * 8 + i]);
          tmp[8 + i] = f2bf(X[ks * 32 + 16 + hi * 8 + i]);
        }
        Af[ks] = pack16(tmp);
      }
    } else {
      const unsigned short* X = (const unsigned short*)Xv + ((size_t)t * 16 + col) * 64;
#pragma unroll
      for (int ks = 0; ks < 2; ++ks) Af[ks] = load_a16frag(X + ks * 32 + hi * 8);
    }
    v8f acc[4];
#pragma unroll
    for (int nt = 0; nt < 4; ++nt)
#pragma unroll
      for (int r = 0; r < 8; ++r) acc[nt][r] = bc[nt];
#pragma unroll
    for (int ks = 0; ks < 2; ++ks)
#pragma unroll
      for (int nt = 0; nt < 4; ++nt) acc[nt] = wmma_bf16(Af[ks], Bf[ks][nt], acc[nt]);
#pragma unroll
    for (int nt = 0; nt < 4; ++nt)
#pragma unroll
      for (int r = 0; r < 8; ++r) {
        size_t idx = ((size_t)t * 16 + hi * 8 + r) * 64 + nt * 16 + col;
        if (Yf)  Yf[idx]  = acc[nt][r];
        if (Ybf) Ybf[idx] = f2bf(acc[nt][r]);
      }
  }
}

// -------- ctx[b] = softmax_L(k)^T @ v, K=512; transposed LDS staging; ctx stored ^T
__global__ __launch_bounds__(512)
void k_ctx(const unsigned short* __restrict__ ksrc, const unsigned short* __restrict__ vsrc,
           const unsigned short* __restrict__ kdst, const unsigned short* __restrict__ vdst,
           unsigned short* __restrict__ ctx0, unsigned short* __restrict__ ctx1) {
  int dir = blockIdx.x >> 8;
  int b = blockIdx.x & 255;
  const unsigned short* kb = (dir == 0 ? kdst : ksrc) + (size_t)b * LL * FF;
  const unsigned short* vb = (dir == 0 ? vdst : vsrc) + (size_t)b * LL * FF;
  unsigned short* co = (dir == 0 ? ctx0 : ctx1) + (size_t)b * FF * FF;

  __shared__ float pm2[64][64];                   // per-rowgroup partials
  __shared__ float partred[8][64];
  __shared__ float colmax[64];
  __shared__ float colinv[64];
  __shared__ __align__(16) unsigned short kslT[64][136];  // [d][l] chunk of softmaxed k^T
  __shared__ __align__(16) unsigned short vT[64][136];    // [e][l] chunk of v^T

  int tid = threadIdx.x;
  int dg = (tid & 7) * 8;                         // 8-column group (16B vector loads)
  int rp = tid >> 3;                              // row group: rows rp*8 .. rp*8+7
  {                                               // pass 1: column max
    float m[8];
#pragma unroll
    for (int i = 0; i < 8; ++i) m[i] = -3.0e38f;
    for (int j = 0; j < 8; ++j) {
      u16x8 kq = *(const u16x8*)(kb + (size_t)(rp * 8 + j) * 64 + dg);
#pragma unroll
      for (int i = 0; i < 8; ++i) m[i] = fmaxf(m[i], bf2f(kq[i]));
    }
#pragma unroll
    for (int i = 0; i < 8; ++i) pm2[rp][dg + i] = m[i];
  }
  __syncthreads();
  {
    int c2 = tid & 63, p2 = tid >> 6;
    float mm = pm2[p2 * 8][c2];
    for (int q = 1; q < 8; ++q) mm = fmaxf(mm, pm2[p2 * 8 + q][c2]);
    partred[p2][c2] = mm;
  }
  __syncthreads();
  if (tid < 64) {
    float mm = partred[0][tid];
    for (int q = 1; q < 8; ++q) mm = fmaxf(mm, partred[q][tid]);
    colmax[tid] = mm;
  }
  __syncthreads();
  float cmv[8];
#pragma unroll
  for (int i = 0; i < 8; ++i) cmv[i] = colmax[dg + i];
  {                                               // pass 2: column sum of exp
    float s[8];
#pragma unroll
    for (int i = 0; i < 8; ++i) s[i] = 0.f;
    for (int j = 0; j < 8; ++j) {
      u16x8 kq = *(const u16x8*)(kb + (size_t)(rp * 8 + j) * 64 + dg);
#pragma unroll
      for (int i = 0; i < 8; ++i) s[i] += __expf(bf2f(kq[i]) - cmv[i]);
    }
#pragma unroll
    for (int i = 0; i < 8; ++i) pm2[rp][dg + i] = s[i];
  }
  __syncthreads();
  {
    int c2 = tid & 63, p2 = tid >> 6;
    float ss = pm2[p2 * 8][c2];
    for (int q = 1; q < 8; ++q) ss += pm2[p2 * 8 + q][c2];
    partred[p2][c2] = ss;
  }
  __syncthreads();
  if (tid < 64) {
    float ss = partred[0][tid];
    for (int q = 1; q < 8; ++q) ss += partred[q][tid];
    colinv[tid] = __builtin_amdgcn_rcpf(ss);      // reciprocal once per column
  }
  __syncthreads();
  float civ[8];
#pragma unroll
  for (int i = 0; i < 8; ++i) civ[i] = colinv[dg + i];

  int lane = tid & 31, wid = tid >> 5;            // 16 waves = 16 C-tiles (4x4)
  int col = lane & 15, hi = lane >> 4;
  int mt = wid >> 2, nt = wid & 3;
  v8f acc;
#pragma unroll
  for (int r = 0; r < 8; ++r) acc[r] = 0.f;

  for (int ch = 0; ch < 4; ++ch) {                // 4 chunks of 128 rows, K=512 total
    int r0 = ch * 128;
    for (int i = tid; i < 1024; i += 512) {       // stage transposed, 16B global loads
      int l = i >> 3;                             // (i&7)*8 == dg for this thread
      u16x8 kq = *(const u16x8*)(kb + (size_t)(r0 + l) * 64 + dg);
      u16x8 vq = *(const u16x8*)(vb + (size_t)(r0 + l) * 64 + dg);
#pragma unroll
      for (int j = 0; j < 8; ++j) {
        kslT[dg + j][l] = f2bf(__expf(bf2f(kq[j]) - cmv[j]) * civ[j]);
        vT[dg + j][l] = vq[j];
      }
    }
    __syncthreads();
#pragma unroll
    for (int ks = 0; ks < 4; ++ks) {
      v16bf A  = load_a16frag(&kslT[mt * 16 + col][ks * 32 + hi * 8]);
      v16bf Bv = load_b16frag(&vT[nt * 16 + col][ks * 32 + hi * 8]);
      acc = wmma_bf16(A, Bv, acc);
    }
    __syncthreads();
  }
  // store ctx TRANSPOSED: coT[e][d]
#pragma unroll
  for (int r = 0; r < 8; ++r)
    co[(size_t)(nt * 16 + col) * 64 + mt * 16 + hi * 8 + r] = f2bf(acc[r]);
}

// ------- out = LN( feat + (softmax(q)*8^-1 @ ctx) @ Wo + bo ); async-DMA LDS staging
__global__ __launch_bounds__(256)
void k_out(const unsigned short* __restrict__ qsrc, const unsigned short* __restrict__ qdst,
           const float* __restrict__ fsrc, const float* __restrict__ fdst,
           const unsigned short* __restrict__ ctx0, const unsigned short* __restrict__ ctx1,
           const unsigned short* __restrict__ woT,
           const float* __restrict__ bo, const float* __restrict__ lng,
           const float* __restrict__ lnb, float* __restrict__ out) {
  int dir = blockIdx.x >> 8;
  int b = blockIdx.x & 255;
  const unsigned short* qb = (dir == 0 ? qsrc : qdst) + (size_t)b * LL * FF;
  const float* xb          = (dir == 0 ? fsrc : fdst) + (size_t)b * LL * FF;
  const unsigned short* cb = (dir == 0 ? ctx0 : ctx1) + (size_t)b * FF * FF;
  float* ob = out + (size_t)dir * BB * LL * FF + (size_t)b * LL * FF;

  __shared__ __align__(16) unsigned short ctxTs[64][64];  // ctx^T [e][d]
  __shared__ __align__(16) unsigned short wos[64][64];    // Wo^T  [n][k]
  __shared__ __align__(16) float t1s[8][16][68];
  __shared__ __align__(16) float bos[64];
  __shared__ __align__(16) float lgs[64];
  __shared__ __align__(16) float lbs[64];
  int tid = threadIdx.x;
  // straight global->LDS copies: use the CDNA5 async DMA path
  for (int i = tid; i < 512; i += 256) {                  // 512 x 16B per array
    const unsigned short* s1 = cb + i * 8;
    unsigned d1 = (unsigned)(size_t)(&ctxTs[0][0]) + i * 16;
    asm volatile("global_load_async_to_lds_b128 %0, %1, off" : : "v"(d1), "v"(s1) : "memory");
    const unsigned short* s2 = woT + i * 8;
    unsigned d2 = (unsigned)(size_t)(&wos[0][0]) + i * 16;
    asm volatile("global_load_async_to_lds_b128 %0, %1, off" : : "v"(d2), "v"(s2) : "memory");
  }
  if (tid < 48) {                                         // bo / ln_g / ln_b, 16B chunks
    int sel = tid >> 4, j = tid & 15;
    const float* sp = (sel == 0 ? bo : sel == 1 ? lng : lnb) + j * 4;
    float* dp = (sel == 0 ? bos : sel == 1 ? lgs : lbs) + j * 4;
    unsigned dd = (unsigned)(size_t)dp;
    asm volatile("global_load_async_to_lds_b128 %0, %1, off" : : "v"(dd), "v"(sp) : "memory");
  }
  asm volatile("s_wait_asynccnt 0x0" ::: "memory");
  __syncthreads();

  int lane = tid & 31, wid = tid >> 5;
  int col = lane & 15, hi = lane >> 4;
  for (int t = wid; t < 32; t += 8) {
    const unsigned short* qr = qb + (size_t)(t * 16 + col) * 64;
    u16x8 rq[8];
#pragma unroll
    for (int v = 0; v < 8; ++v) rq[v] = *(const u16x8*)(qr + v * 8);
    float mx = -3.0e38f;
#pragma unroll
    for (int v = 0; v < 8; ++v)
#pragma unroll
      for (int i = 0; i < 8; ++i) mx = fmaxf(mx, bf2f(rq[v][i]));
    float sum = 0.f;
#pragma unroll
    for (int v = 0; v < 8; ++v)
#pragma unroll
      for (int i = 0; i < 8; ++i) sum += __expf(bf2f(rq[v][i]) - mx);
    float scale = 0.125f * __builtin_amdgcn_rcpf(sum);    // softmax * 64^-0.5

    v8f acc1[4];
#pragma unroll
    for (int nt = 0; nt < 4; ++nt)
#pragma unroll
      for (int r = 0; r < 8; ++r) acc1[nt][r] = 0.f;
    unsigned short tmp[16];
#pragma unroll
    for (int ks = 0; ks < 2; ++ks) {              // attn = qs @ ctx
      u16x8 vlo = hi ? rq[ks * 4 + 1] : rq[ks * 4 + 0];
      u16x8 vhi = hi ? rq[ks * 4 + 3] : rq[ks * 4 + 2];
#pragma unroll
      for (int i = 0; i < 8; ++i) {
        tmp[i]     = f2bf(__expf(bf2f(vlo[i]) - mx) * scale);
        tmp[8 + i] = f2bf(__expf(bf2f(vhi[i]) - mx) * scale);
      }
      v16bf A = pack16(tmp);
#pragma unroll
      for (int nt = 0; nt < 4; ++nt)
        acc1[nt] = wmma_bf16(A, load_b16frag(&ctxTs[nt * 16 + col][ks * 32 + hi * 8]),
                             acc1[nt]);
    }
    // stage T1 in LDS to re-layout C->A between the chained GEMMs (same-wave DS order)
#pragma unroll
    for (int nt = 0; nt < 4; ++nt)
#pragma unroll
      for (int r = 0; r < 8; ++r) t1s[wid][hi * 8 + r][nt * 16 + col] = acc1[nt][r];

    v8f acc2[4];
#pragma unroll
    for (int nt = 0; nt < 4; ++nt)
#pragma unroll
      for (int r = 0; r < 8; ++r) acc2[nt][r] = 0.f;
#pragma unroll
    for (int ks = 0; ks < 2; ++ks) {              // T2 = T1 @ Wo
      const float* tp = &t1s[wid][col][ks * 32 + hi * 8];
      f32x4 x0 = *(const f32x4*)tp;
      f32x4 x1 = *(const f32x4*)(tp + 4);
      f32x4 y0 = *(const f32x4*)(tp + 16);
      f32x4 y1 = *(const f32x4*)(tp + 20);
#pragma unroll
      for (int i = 0; i < 4; ++i) {
        tmp[i]      = f2bf(x0[i]);
        tmp[4 + i]  = f2bf(x1[i]);
        tmp[8 + i]  = f2bf(y0[i]);
        tmp[12 + i] = f2bf(y1[i]);
      }
      v16bf A2 = pack16(tmp);
#pragma unroll
      for (int nt = 0; nt < 4; ++nt)
        acc2[nt] = wmma_bf16(A2, load_b16frag(&wos[nt * 16 + col][ks * 32 + hi * 8]),
                             acc2[nt]);
    }
    float vals[4][8];
#pragma unroll
    for (int nt = 0; nt < 4; ++nt)
#pragma unroll
      for (int r = 0; r < 8; ++r)
        vals[nt][r] = acc2[nt][r] + xb[(size_t)(t * 16 + hi * 8 + r) * 64 + nt * 16 + col]
                    + bos[nt * 16 + col];
    // LayerNorm: reduce over the 16 lanes (xor 1,2,4,8 keeps hi fixed) x 4 ntiles
#pragma unroll
    for (int r = 0; r < 8; ++r) {
      float s1 = vals[0][r] + vals[1][r] + vals[2][r] + vals[3][r];
      float s2 = vals[0][r] * vals[0][r] + vals[1][r] * vals[1][r]
               + vals[2][r] * vals[2][r] + vals[3][r] * vals[3][r];
#pragma unroll
      for (int msk = 1; msk <= 8; msk <<= 1) {
        s1 += __shfl_xor(s1, msk, 32);
        s2 += __shfl_xor(s2, msk, 32);
      }
      float mu = s1 * 0.015625f;
      float var = s2 * 0.015625f - mu * mu;
      float rs = rsqrtf(var + 1e-5f);
#pragma unroll
      for (int nt = 0; nt < 4; ++nt) {
        int cc = nt * 16 + col;
        ob[(size_t)(t * 16 + hi * 8 + r) * 64 + cc] = (vals[nt][r] - mu) * rs * lgs[cc] + lbs[cc];
      }
    }
  }
}

extern "C" void kernel_launch(void* const* d_in, const int* in_sizes, int n_in,
                              void* d_out, int out_size, void* d_ws, size_t ws_size,
                              hipStream_t stream) {
  (void)in_sizes; (void)n_in; (void)out_size; (void)ws_size;
  const int*   src_ids = (const int*)d_in[0];
  const int*   dst_ids = (const int*)d_in[1];
  const float* enc_w1  = (const float*)d_in[2];
  const float* enc_b1  = (const float*)d_in[3];
  const float* enc_w2  = (const float*)d_in[4];
  const float* enc_b2  = (const float*)d_in[5];
  const float* Wq      = (const float*)d_in[6];
  const float* Wk      = (const float*)d_in[7];
  const float* Wv      = (const float*)d_in[8];
  const float* Wo      = (const float*)d_in[9];
  const float* bo      = (const float*)d_in[10];
  const float* ln_g    = (const float*)d_in[11];
  const float* ln_b    = (const float*)d_in[12];
  float* out = (float*)d_out;

  const size_t MF = (size_t)BB * LL * FF;          // 8,388,608 elements
  unsigned short* g0 = (unsigned short*)d_ws;      // g_src bf16, later reused as q_src bf16
  unsigned short* g1 = g0 + MF;                    // g_dst / q_dst
  float* f0 = (float*)(g1 + MF);                   // feat_src f32 (residual)
  float* f1 = f0 + MF;                             // feat_dst f32
  unsigned short* k0 = (unsigned short*)(f1 + MF); // k_src bf16
  unsigned short* k1 = k0 + MF;                    // k_dst
  unsigned short* v0 = k1 + MF;                    // v_src
  unsigned short* v1 = v0 + MF;                    // v_dst
  unsigned short* c0 = v1 + MF;                    // ctx^T for src_out [B,64,64]
  unsigned short* c1 = c0 + (size_t)BB * FF * FF;  // ctx^T for dst_out
  unsigned short* wb = c1 + (size_t)BB * FF * FF;  // bf16 W^T: w2,Wq,Wk,Wv,Wo
  unsigned short* w2b = wb;
  unsigned short* wqb = wb + 4096;
  unsigned short* wkb = wb + 8192;
  unsigned short* wvb = wb + 12288;
  unsigned short* wob = wb + 16384;

  k_cvt_weights<<<16, 256, 0, stream>>>(enc_w2, Wq, Wk, Wv, Wo, wb);
  k_count_encode<<<BB, 256, 0, stream>>>(src_ids, dst_ids, enc_w1, enc_b1, g0, g1);
  // feat = g @ w2 + 2*b2   (f32 out for residual)
  k_gemm64<false><<<128, 256, 0, stream>>>(g0, w2b, enc_b2, 2.0f, f0, nullptr);
  k_gemm64<false><<<128, 256, 0, stream>>>(g1, w2b, enc_b2, 2.0f, f1, nullptr);
  // q/k/v projections (bf16 out); q reuses the g buffers
  k_gemm64<true ><<<128, 256, 0, stream>>>(f0, wqb, nullptr, 0.f, nullptr, g0);
  k_gemm64<true ><<<128, 256, 0, stream>>>(f1, wqb, nullptr, 0.f, nullptr, g1);
  k_gemm64<true ><<<128, 256, 0, stream>>>(f0, wkb, nullptr, 0.f, nullptr, k0);
  k_gemm64<true ><<<128, 256, 0, stream>>>(f1, wkb, nullptr, 0.f, nullptr, k1);
  k_gemm64<true ><<<128, 256, 0, stream>>>(f0, wvb, nullptr, 0.f, nullptr, v0);
  k_gemm64<true ><<<128, 256, 0, stream>>>(f1, wvb, nullptr, 0.f, nullptr, v1);
  // ctx^T = (softmax_L(k)^T @ v)^T for both directions (dir0 uses k_dst/v_dst)
  k_ctx<<<2 * BB, 512, 0, stream>>>(k0, v0, k1, v1, c0, c1);
  // final: softmax(q) @ ctx @ Wo + residual + LayerNorm
  k_out<<<2 * BB, 256, 0, stream>>>(g0, g1, f0, f1, c0, c1, wob, bo, ln_g, ln_b, out);
}
